// CaptionModel_16552803959405
// MI455X (gfx1250) — matrix operand order repaired and checked by reading, not compiled, not run on hardware
//
#include <hip/hip_runtime.h>

// ---------------------------------------------------------------------------
// CDNA5 (gfx1250) GRU caption decoder.
// B=256, H=512, VOCAB=100, STEPS=200. wave32, v_wmma_f32_16x16x32_bf16.
//
// Launch-latency-optimized: projection of h_t is fused (as extra blocks) into
// the kernel that computes step t+1, halving the serial launch chain.
// ---------------------------------------------------------------------------

typedef __bf16 bf16_t;
typedef __attribute__((ext_vector_type(16))) __bf16 v16bf;
typedef __attribute__((ext_vector_type(8)))  __bf16 v8bf;
typedef __attribute__((ext_vector_type(8)))  float  v8f;

#define GRU_H   512
#define GRU_B   256
#define GRU_V   100
#define GRU_VP  112   // vocab padded to 7 tiles of 16
#define GRU_T   200

#define STEP_BLOCKS 64   // (256/16)*(512/16) waves / 8 waves-per-block
#define PROJ_BLOCKS 14   // (256/16)*(112/16) waves / 8

__device__ __forceinline__ float sigmoidf_(float x) {
    return 1.0f / (1.0f + __expf(-x));
}

// --- A tile (16x32 bf16, M x K) in contiguous k order, as B ----------------------
// lanes 0-15: M=lane, K=0..15 across the 8 VGPRs
// lanes 16-31: M=lane-16, K=16..31
__device__ __forceinline__ v16bf load_a_tile(const bf16_t* __restrict__ base,
                                             int m0, int k0, int lane) {
    const int m  = lane & 15;
    const int kb = (lane >> 4) << 4;   // 0 or 16
    const bf16_t* p = base + (size_t)(m0 + m) * GRU_H + k0 + kb;
    v8bf lo = *(const v8bf*)(p);        // K = kb .. kb+7
    v8bf hi = *(const v8bf*)(p + 8);    // K = kb+8 .. kb+15
    v16bf a;
#pragma unroll
    for (int i = 0; i < 8; ++i) { a[i] = lo[i]; a[i + 8] = hi[i]; }
    return a;
}

// --- B tile (32x16 bf16, K x N) from row-major [N][K] weight ---------------
// lanes 0-15: N=lane, K=0..15 across the 8 VGPRs; lanes 16-31: N=lane-16, K=16..31
__device__ __forceinline__ v16bf load_b_tile(const bf16_t* __restrict__ W,
                                             int n0, int k0, int lane, int ldk) {
    const int n  = lane & 15;
    const int kh = (lane >> 4) << 4;   // 0 or 16
    return *(const v16bf*)(W + (size_t)(n0 + n) * ldk + k0 + kh);
}

// --- projection of one h snapshot into out[:, :, t] (device helper) --------
__device__ __forceinline__ void proj_tile(const bf16_t* __restrict__ h_bf,
                                          const bf16_t* __restrict__ Wp,
                                          const float* __restrict__ pb,
                                          float* __restrict__ out,
                                          int wg, int lane, int t) {
    const int m0 = (wg & 15) << 4;
    const int n0 = (wg >> 4) << 4;
    v8f acc = {};
    for (int k0 = 0; k0 < GRU_H; k0 += 32) {
        v16bf a = load_a_tile(h_bf, m0, k0, lane);
        v16bf b = load_b_tile(Wp, n0, k0, lane, GRU_H);
        acc = __builtin_amdgcn_wmma_f32_16x16x32_bf16(false, a, false, b, (short)0, acc, false, false);
    }
    const int v  = n0 + (lane & 15);
    const int mh = (lane >> 4) << 3;
    if (v < GRU_V) {
        const float bias = pb[v];
#pragma unroll
        for (int g = 0; g < 8; ++g) {
            const int b_row = m0 + mh + g;
            out[(size_t)b_row * (GRU_V * GRU_T) + (size_t)v * GRU_T + t] = acc[g] + bias;
        }
    }
}

// ---------------------------------------------------------------------------
// Prep kernels (run once)
// ---------------------------------------------------------------------------

// Combine gate weights (r/z use W_ih+W_hh since x==h for t>=1) -> bf16.
__global__ __launch_bounds__(256) void prep_weights_kernel(
    const float* __restrict__ w_ih, const float* __restrict__ w_hh,
    bf16_t* __restrict__ Wr,  bf16_t* __restrict__ Wz,
    bf16_t* __restrict__ Wni, bf16_t* __restrict__ Wnh,
    bf16_t* __restrict__ Whr, bf16_t* __restrict__ Whz) {
    int idx = blockIdx.x * blockDim.x + threadIdx.x;        // over 512*512
    if (idx >= GRU_H * GRU_H) return;
    int j = idx / GRU_H, k = idx % GRU_H;
    size_t r_off = (size_t)j * GRU_H + k;
    size_t z_off = (size_t)(GRU_H + j) * GRU_H + k;
    size_t n_off = (size_t)(2 * GRU_H + j) * GRU_H + k;
    Wr[idx]  = (bf16_t)(w_ih[r_off] + w_hh[r_off]);
    Wz[idx]  = (bf16_t)(w_ih[z_off] + w_hh[z_off]);
    Wni[idx] = (bf16_t)(w_ih[n_off]);
    Wnh[idx] = (bf16_t)(w_hh[n_off]);
    Whr[idx] = (bf16_t)(w_hh[r_off]);   // step-0 variants (x = embed(SOS))
    Whz[idx] = (bf16_t)(w_hh[z_off]);
}

// Projection weight -> bf16, padded [112][512] (pad rows are zero).
__global__ __launch_bounds__(256) void prep_proj_kernel(
    const float* __restrict__ proj_w, bf16_t* __restrict__ Wp) {
    int idx = blockIdx.x * blockDim.x + threadIdx.x;        // over 112*512
    if (idx >= GRU_VP * GRU_H) return;
    int v = idx / GRU_H, k = idx % GRU_H;
    Wp[idx] = (v < GRU_V) ? (bf16_t)proj_w[(size_t)v * GRU_H + k] : (bf16_t)0.0f;
}

// gi0[j] = embed[SOS] . w_ih[j] + b_ih[j]  (x of step 0 is batch-uniform),
// plus combined biases for the steady-state steps.
__global__ __launch_bounds__(256) void prep_bias_kernel(
    const float* __restrict__ embed, const float* __restrict__ w_ih,
    const float* __restrict__ b_ih,  const float* __restrict__ b_hh,
    float* __restrict__ brz, float* __restrict__ b0rz,
    float* __restrict__ bni, float* __restrict__ bnh, float* __restrict__ b0ni) {
    int j = blockIdx.x * blockDim.x + threadIdx.x;          // over 1536
    if (j >= 3 * GRU_H) return;
    float acc = b_ih[j];
    const float* wr = w_ih + (size_t)j * GRU_H;
    for (int k = 0; k < GRU_H; ++k) acc += embed[k] * wr[k];  // SOS_IDX == 0 (row 0)
    if (j < 2 * GRU_H) {
        brz[j]  = b_ih[j] + b_hh[j];   // steady state: x==h -> biases combine
        b0rz[j] = acc + b_hh[j];       // step 0: gi0 already holds b_ih
    } else {
        int jj = j - 2 * GRU_H;
        bni[jj]  = b_ih[j];
        bnh[jj]  = b_hh[j];
        b0ni[jj] = acc;                // step 0: i_n is a constant per column
    }
}

// h0 = feat, in f32 (master) and bf16 (GEMM operand).
__global__ __launch_bounds__(256) void init_h_kernel(
    const float* __restrict__ feat, float* __restrict__ hf, bf16_t* __restrict__ hb) {
    int idx = blockIdx.x * blockDim.x + threadIdx.x;        // over 256*512
    if (idx >= GRU_B * GRU_H) return;
    float v = feat[idx];
    hf[idx] = v;
    hb[idx] = (bf16_t)v;
}

// ---------------------------------------------------------------------------
// Fused step kernel:
//   blocks [0, 64):  GRU step  h_bf/h_f32 -> ho_bf/ho_f32 (one 16x16 tile/wave)
//   blocks [64, 78): projection of the *input* h (the previous step's output,
//                    already complete) into out[:, :, t_prev]  [DO_PROJ only]
// Block-level branch is uniform, so EXEC is all-ones around every WMMA.
// ---------------------------------------------------------------------------
template <bool FIRST, bool DO_PROJ>
__global__ __launch_bounds__(256) void gru_fused_kernel(
    const bf16_t* __restrict__ h_bf, const float* __restrict__ h_f32,
    const bf16_t* __restrict__ Wr,  const bf16_t* __restrict__ Wz,
    const bf16_t* __restrict__ Wni, const bf16_t* __restrict__ Wnh,
    const float* __restrict__ bias_rz, const float* __restrict__ bias_ni,
    const float* __restrict__ bias_nh,
    bf16_t* __restrict__ ho_bf, float* __restrict__ ho_f32,
    const bf16_t* __restrict__ Wp, const float* __restrict__ pb,
    float* __restrict__ out, int t_prev) {
    const int lane = threadIdx.x & 31;

    if constexpr (DO_PROJ) {
        if (blockIdx.x >= STEP_BLOCKS) {
            const int wg = (blockIdx.x - STEP_BLOCKS) * 8 + (threadIdx.x >> 5); // 0..111
            proj_tile(h_bf, Wp, pb, out, wg, lane, t_prev);
            return;
        }
    }

    const int wg = blockIdx.x * 8 + (threadIdx.x >> 5);     // 0..511
    const int m0 = (wg & 15) << 4;                          // batch-row tile
    const int n0 = (wg >> 4) << 4;                          // hidden-col tile

    v8f ar = {}, az = {}, ani = {}, anh = {};
    for (int k0 = 0; k0 < GRU_H; k0 += 32) {
        v16bf a  = load_a_tile(h_bf, m0, k0, lane);
        v16bf br = load_b_tile(Wr,  n0, k0, lane, GRU_H);
        ar = __builtin_amdgcn_wmma_f32_16x16x32_bf16(false, a, false, br, (short)0, ar, false, false);
        v16bf bz = load_b_tile(Wz,  n0, k0, lane, GRU_H);
        az = __builtin_amdgcn_wmma_f32_16x16x32_bf16(false, a, false, bz, (short)0, az, false, false);
        if constexpr (!FIRST) {
            v16bf bi = load_b_tile(Wni, n0, k0, lane, GRU_H);
            ani = __builtin_amdgcn_wmma_f32_16x16x32_bf16(false, a, false, bi, (short)0, ani, false, false);
        }
        v16bf bh = load_b_tile(Wnh, n0, k0, lane, GRU_H);
        anh = __builtin_amdgcn_wmma_f32_16x16x32_bf16(false, a, false, bh, (short)0, anh, false, false);
    }

    const int col = n0 + (lane & 15);
    const int mh  = (lane >> 4) << 3;   // C/D layout: lanes 16-31 hold M+8
    const float brv  = bias_rz[col];
    const float bzv  = bias_rz[GRU_H + col];
    const float bniv = bias_ni[col];
    const float bnhv = bias_nh[col];
#pragma unroll
    for (int g = 0; g < 8; ++g) {
        const int row = m0 + mh + g;
        float r = sigmoidf_(ar[g] + brv);
        float z = sigmoidf_(az[g] + bzv);
        float hn_pre = anh[g] + bnhv;
        float nn;
        if constexpr (FIRST) nn = tanhf(bniv + r * hn_pre);          // i_n constant
        else                 nn = tanhf(ani[g] + bniv + r * hn_pre);
        float hold = h_f32[(size_t)row * GRU_H + col];
        float hv   = (1.0f - z) * nn + z * hold;
        ho_f32[(size_t)row * GRU_H + col] = hv;
        ho_bf [(size_t)row * GRU_H + col] = (bf16_t)hv;
    }
}

// Stand-alone projection for the final step's hidden state.
__global__ __launch_bounds__(256) void proj_step_kernel(
    const bf16_t* __restrict__ h_bf, const bf16_t* __restrict__ Wp,
    const float* __restrict__ pb, float* __restrict__ out, int t) {
    const int lane = threadIdx.x & 31;
    const int wg   = blockIdx.x * 8 + (threadIdx.x >> 5);   // 0..111
    proj_tile(h_bf, Wp, pb, out, wg, lane, t);
}

// ---------------------------------------------------------------------------
// Host launcher
// ---------------------------------------------------------------------------
extern "C" void kernel_launch(void* const* d_in, const int* in_sizes, int n_in,
                              void* d_out, int out_size, void* d_ws, size_t ws_size,
                              hipStream_t stream) {
    (void)in_sizes; (void)n_in; (void)out_size; (void)ws_size;
    const float* feat   = (const float*)d_in[0];   // [256,512]
    const float* embed  = (const float*)d_in[1];   // [100,512]
    const float* w_ih   = (const float*)d_in[2];   // [1536,512]
    const float* w_hh   = (const float*)d_in[3];   // [1536,512]
    const float* b_ih   = (const float*)d_in[4];   // [1536]
    const float* b_hh   = (const float*)d_in[5];   // [1536]
    const float* proj_w = (const float*)d_in[6];   // [100,512]
    const float* proj_b = (const float*)d_in[7];   // [100]
    float* out = (float*)d_out;                    // [256,100,200]

    // --- carve workspace (aligned 256B) ---
    char* ws = (char*)d_ws;
    size_t off = 0;
    auto carve = [&](size_t bytes) -> char* {
        char* p = ws + off;
        off = (off + bytes + 255) & ~(size_t)255;
        return p;
    };
    const size_t WBLK = (size_t)GRU_H * GRU_H * sizeof(bf16_t);   // 512KB
    bf16_t* Wr   = (bf16_t*)carve(WBLK);
    bf16_t* Wz   = (bf16_t*)carve(WBLK);
    bf16_t* Wni  = (bf16_t*)carve(WBLK);
    bf16_t* Wnh  = (bf16_t*)carve(WBLK);
    bf16_t* Whr  = (bf16_t*)carve(WBLK);
    bf16_t* Whz  = (bf16_t*)carve(WBLK);
    bf16_t* Wp   = (bf16_t*)carve((size_t)GRU_VP * GRU_H * sizeof(bf16_t));
    float*  brz  = (float*)carve(2 * GRU_H * sizeof(float));
    float*  b0rz = (float*)carve(2 * GRU_H * sizeof(float));
    float*  bni  = (float*)carve(GRU_H * sizeof(float));
    float*  bnh  = (float*)carve(GRU_H * sizeof(float));
    float*  b0ni = (float*)carve(GRU_H * sizeof(float));
    float*  hf[2];
    bf16_t* hb[2];
    hf[0] = (float*) carve((size_t)GRU_B * GRU_H * sizeof(float));
    hf[1] = (float*) carve((size_t)GRU_B * GRU_H * sizeof(float));
    hb[0] = (bf16_t*)carve((size_t)GRU_B * GRU_H * sizeof(bf16_t));
    hb[1] = (bf16_t*)carve((size_t)GRU_B * GRU_H * sizeof(bf16_t));

    // --- prep (once, deterministic) ---
    prep_weights_kernel<<<(GRU_H * GRU_H + 255) / 256, 256, 0, stream>>>(
        w_ih, w_hh, Wr, Wz, Wni, Wnh, Whr, Whz);
    prep_proj_kernel<<<(GRU_VP * GRU_H + 255) / 256, 256, 0, stream>>>(proj_w, Wp);
    prep_bias_kernel<<<(3 * GRU_H + 255) / 256, 256, 0, stream>>>(
        embed, w_ih, b_ih, b_hh, brz, b0rz, bni, bnh, b0ni);
    init_h_kernel<<<(GRU_B * GRU_H + 255) / 256, 256, 0, stream>>>(feat, hf[0], hb[0]);

    // --- 200 sequential GRU steps; projection of h_t rides in step t+1 ---
    // t=0: GRU only (nothing to project yet).
    gru_fused_kernel<true, false><<<STEP_BLOCKS, 256, 0, stream>>>(
        hb[0], hf[0], Whr, Whz, Wnh /*unused*/, Wnh,
        b0rz, b0ni, bnh, hb[1], hf[1],
        Wp, proj_b, out, 0);
    // t=1..199: GRU step + projection of the previous step's h (same buffer).
    for (int t = 1; t < GRU_T; ++t) {
        const int cur = t & 1, nxt = cur ^ 1;
        gru_fused_kernel<false, true><<<STEP_BLOCKS + PROJ_BLOCKS, 256, 0, stream>>>(
            hb[cur], hf[cur], Wr, Wz, Wni, Wnh,
            brz, bni, bnh, hb[nxt], hf[nxt],
            Wp, proj_b, out, t - 1);
    }
    // Final projection: h after step 199 lives in buffer GRU_T & 1; hb[last ^ 1] below is step 198's h:
    const int last = GRU_T & 1;   // buffer written by step 199 (cur=199&1=1, nxt=0)
    proj_step_kernel<<<PROJ_BLOCKS, 256, 0, stream>>>(
        hb[last ^ 1], Wp, proj_b, out, GRU_T - 1);
}
